// SelfAttention_44719199486705
// MI455X (gfx1250) — matrix operand order repaired and checked by reading, not compiled, not run on hardware
//
#include <hip/hip_runtime.h>
#include <hip/hip_bf16.h>

// ---------------------------------------------------------------------------
// Self-attention forward for MI455X (gfx1250, wave32, WMMA, async-LDS).
//   Kernel 0: f32 -> bf16 bulk conversion (x, Wq, Wk, Wv) using v_cvt bf16.
//   Kernel 1: QKV projection GEMM, all-bf16 operands straight from global
//             (L2-resident), 4 WMMAs per 32-deep K-step per wave.
//   Kernel 2: flash attention; K/V key-blocks are double-buffered in LDS via
//             GLOBAL_LOAD_ASYNC_TO_LDS_B128 (ASYNCcnt) and shared by all 4
//             waves; 8 WMMAs + online softmax per 32-key step.
// ---------------------------------------------------------------------------

typedef __attribute__((ext_vector_type(16))) __bf16          v16bf;
typedef __attribute__((ext_vector_type(8)))  float           v8f;
typedef __attribute__((ext_vector_type(16))) unsigned short  v16us;
typedef __attribute__((ext_vector_type(8)))  unsigned short  v8us;

#define BB 2
#define SS 2048
#define DDIM 1024
#define HH 16
#define HD 64
#define BSROWS (BB * SS)

__device__ __forceinline__ unsigned short f2bf(float f) {
  __bf16 h = (__bf16)f;                      // hardware cvt path
  return __builtin_bit_cast(unsigned short, h);
}
__device__ __forceinline__ v16bf as_bf(v16us u) {
  return __builtin_bit_cast(v16bf, u);
}

// async copy: 16B per lane, global -> LDS, tracked by ASYNCcnt
__device__ __forceinline__ void async_b128(void* lds_dst, const void* gsrc) {
  asm volatile("global_load_async_to_lds_b128 %0, %1, off"
               :
               : "v"((unsigned)(unsigned long long)lds_dst),
                 "v"((unsigned long long)gsrc)
               : "memory");
}
__device__ __forceinline__ void wait_async0() {
  asm volatile("s_wait_asynccnt 0x0" ::: "memory");
}

// ---------------------------------------------------------------------------
// Kernel 0: bulk f32 -> bf16 (8 elements/thread, vectorized 16B stores)
// ---------------------------------------------------------------------------
__global__ __launch_bounds__(256) void cvt_bf16_kernel(
    const float* __restrict__ src, unsigned short* __restrict__ dst, int n)
{
  int i = (blockIdx.x * 256 + threadIdx.x) * 8;
  if (i >= n) return;
  float4 a = *(const float4*)&src[i];
  float4 b = *(const float4*)&src[i + 4];
  v8us o;
  o[0] = f2bf(a.x); o[1] = f2bf(a.y); o[2] = f2bf(a.z); o[3] = f2bf(a.w);
  o[4] = f2bf(b.x); o[5] = f2bf(b.y); o[6] = f2bf(b.z); o[7] = f2bf(b.w);
  *(v8us*)&dst[i] = o;
}

// ---------------------------------------------------------------------------
// Kernel 1: Y = x @ W^T + b (blockIdx.z selects Q/K/V), all-bf16 operands.
// Block = 256 threads = 8 waves; each wave computes a 16(M) x 64(N) tile
// with 4 f32 accumulators; K-loop over D=1024 in steps of 32.
// ---------------------------------------------------------------------------
__global__ __launch_bounds__(256) void qkv_proj_kernel(
    const unsigned short* __restrict__ xb,      // [BS, D] bf16
    const unsigned short* __restrict__ wqb,     // [D, D] bf16
    const unsigned short* __restrict__ wkb,
    const unsigned short* __restrict__ wvb,
    const float* __restrict__ bq, const float* __restrict__ bk,
    const float* __restrict__ bv,
    unsigned short* __restrict__ qo, unsigned short* __restrict__ ko,
    unsigned short* __restrict__ vo)
{
  const int lane = threadIdx.x & 31;
  const int wave = threadIdx.x >> 5;               // 0..7
  const int m0   = blockIdx.x * 16;                // flat row base (b*S+s)
  const int n0   = blockIdx.y * 512 + wave * 64;   // output-feature base
  const int z    = blockIdx.z;                     // 0=Q 1=K 2=V

  const unsigned short* W = (z == 0) ? wqb : (z == 1) ? wkb : wvb;
  const float* bias       = (z == 0) ? bq  : (z == 1) ? bk  : bv;

  const int col = lane & 15;
  const int hi  = lane >> 4;

  const unsigned short* ar = xb + (size_t)(m0 + col) * DDIM;  // A row (lane)

  v8f acc[4] = {};
  for (int k0 = 0; k0 < DDIM; k0 += 32) {
    // A fragment (contiguous k order, same as B): lane<16 row=col K={0..15},
    // lane>=16 row=col K={16..31} -> two 16B loads
    v8us alo = *(const v8us*)&ar[k0 + hi * 16];
    v8us ahi = *(const v8us*)&ar[k0 + hi * 16 + 8];
    v16bf a = as_bf(__builtin_shufflevector(alo, ahi,
        0,1,2,3,4,5,6,7,8,9,10,11,12,13,14,15));

    // B fragments (32x16): column = W row; lanes 0-15 d=0..15,
    // lanes 16-31 d=16..31 -> one 32B load per N-tile
#pragma unroll
    for (int t = 0; t < 4; ++t) {
      const size_t e = n0 + t * 16 + col;
      v16us bu = *(const v16us*)&W[e * DDIM + k0 + hi * 16];
      acc[t] = __builtin_amdgcn_wmma_f32_16x16x32_bf16(
          false, a, false, as_bf(bu), (short)0, acc[t], false, false);
    }
  }

  // epilogue: bias, bf16 convert, scatter into per-head layouts
  const int mBase = hi * 8;
#pragma unroll
  for (int t = 0; t < 4; ++t) {
    const int e  = n0 + t * 16 + col;
    const int h  = e >> 6;               // head
    const int dd = e & 63;               // dim within head
    const float bsc = bias[e];
#pragma unroll
    for (int r = 0; r < 8; ++r) {
      int ms = m0 + mBase + r;
      int b  = ms >> 11;                 // / S
      int s  = ms & (SS - 1);
      unsigned short y = f2bf(acc[t][r] + bsc);
      if (z == 2) {                      // V^T: [B,H,64,S]
        vo[(((size_t)b * HH + h) * HD + dd) * SS + s] = y;
      } else {                           // Q,K: [B,H,S,64]
        unsigned short* dst = (z == 0) ? qo : ko;
        dst[(((size_t)b * HH + h) * SS + s) * HD + dd] = y;
      }
    }
  }
}

// ---------------------------------------------------------------------------
// Kernel 2: flash attention. Block = 128 threads = 4 waves sharing one (b,h);
// each wave owns a 16-query tile. Key loop in steps of 32: wave 0 async-
// prefetches the next K/V block into the alternate LDS buffer while all
// waves compute the current block from LDS (4 WMMA scores + online softmax
// + 4 WMMA PxV).
// ---------------------------------------------------------------------------
__global__ __launch_bounds__(128) void attn_kernel(
    const unsigned short* __restrict__ qi,
    const unsigned short* __restrict__ ki,
    const unsigned short* __restrict__ vt,
    const int* __restrict__ mask,
    float* __restrict__ out)
{
  const int lane = threadIdx.x & 31;
  const int wave = threadIdx.x >> 5;     // 0..3
  const int h  = blockIdx.y;
  const int b  = blockIdx.z;
  const int q0 = (blockIdx.x * 4 + wave) * 16;

  // double-buffered K/V key-blocks, shared by the whole block (async DMA dst)
  __shared__ alignas(32) unsigned short kbuf[2][32 * HD];  // [key][d]
  __shared__ alignas(32) unsigned short vbuf[2][HD * 32];  // [d][key]
  __shared__ float sc[4][16][33];        // per-wave 16x32 score tile (+pad)
  __shared__ float mrow[4][16];
  __shared__ float lrow[4][16];
  __shared__ float rsc[4][16];

  const int col = lane & 15;
  const int hi  = lane >> 4;
  const size_t bh = (size_t)b * HH + h;
  const unsigned short* qbase = qi + (bh * SS + q0) * HD;
  const char* kbytes = (const char*)(ki + bh * SS * HD);  // 128B per key row
  const char* vbytes = (const char*)(vt + bh * HD * SS);  // 4096B per d row

  // Q fragments: two 16x32 k-steps covering hd=64 (A-matrix layout)
  v16bf aq[2];
#pragma unroll
  for (int f = 0; f < 2; ++f) {
    const unsigned short* qr = qbase + (size_t)col * HD + f * 32;
    v8us lo  = *(const v8us*)&qr[hi * 16];
    v8us hi8 = *(const v8us*)&qr[hi * 16 + 8];
    aq[f] = as_bf(__builtin_shufflevector(lo, hi8,
        0,1,2,3,4,5,6,7,8,9,10,11,12,13,14,15));
  }

  float maskadd = 0.0f;
  if (lane < 16) {
    mrow[wave][lane] = -INFINITY;
    lrow[wave][lane] = 0.0f;
    // reference: mask indexes the QUERY axis ([B,1,S,1] broadcast)
    maskadd = (1.0f - (float)mask[b * SS + q0 + lane]) * -1e9f;
  }

  v8f acc[4] = {};

  // ---- issue async loads for key-block `kk` into buffer `nb` (wave 0 only)
  auto issue_block = [&](int kk, int nb) {
    // K tile: keys kk..kk+31 are 32 contiguous 128B rows -> 4KB contiguous
    const char* ks = kbytes + (size_t)kk * (HD * 2);
    char*       kd = (char*)&kbuf[nb][0];
#pragma unroll
    for (int j = 0; j < 8; ++j) {
      unsigned off = (unsigned)(j * 32 + lane) * 16;
      async_b128(kd + off, ks + off);
    }
    // V^T tile: 64 rows (d) x 64B, global row stride 4096B
#pragma unroll
    for (int j = 0; j < 8; ++j) {
      int slot = j * 32 + lane;          // 0..255
      int row  = slot >> 2;              // d
      int wi   = (slot & 3) * 16;        // 16B quadrant within 64B row
      async_b128((char*)&vbuf[nb][0] + row * 64 + wi,
                 vbytes + (size_t)row * (SS * 2) + (size_t)kk * 2 + wi);
    }
  };

  if (wave == 0) issue_block(0, 0);

  for (int kk = 0; kk < SS; kk += 32) {
    const int buf = (kk >> 5) & 1;
    if (wave == 0) wait_async0();        // current buffer landed in LDS
    __syncthreads();                      // publish to all waves
    if (wave == 0 && kk + 32 < SS) issue_block(kk + 32, buf ^ 1);

    // ---- scores: two 16x16 tiles over keys [kk, kk+32), K from LDS
#pragma unroll
    for (int t = 0; t < 2; ++t) {
      const unsigned short* kr = &kbuf[buf][(t * 16 + col) * HD];
      v8f s = {};
#pragma unroll
      for (int f = 0; f < 2; ++f) {      // k-steps over hd
        v16us bu = *(const v16us*)&kr[f * 32 + hi * 16];
        s = __builtin_amdgcn_wmma_f32_16x16x32_bf16(
            false, aq[f], false, as_bf(bu), (short)0, s, false, false);
      }
#pragma unroll
      for (int r = 0; r < 8; ++r)
        sc[wave][hi * 8 + r][t * 16 + col] = s[r];
    }

    // ---- online softmax: lanes 0..15 each own one query row
    if (lane < 16) {
      float mx = -INFINITY;
#pragma unroll
      for (int j = 0; j < 32; ++j) {
        float v = sc[wave][lane][j] * 0.125f + maskadd;  // 1/sqrt(64)
        sc[wave][lane][j] = v;
        mx = fmaxf(mx, v);
      }
      float mold = mrow[wave][lane];
      float mnew = fmaxf(mold, mx);
      float rs   = __expf(mold - mnew);
      float sum  = 0.0f;
#pragma unroll
      for (int j = 0; j < 32; ++j) {
        float p = __expf(sc[wave][lane][j] - mnew);
        sc[wave][lane][j] = p;
        sum += p;
      }
      mrow[wave][lane] = mnew;
      lrow[wave][lane] = lrow[wave][lane] * rs + sum;
      rsc[wave][lane]  = rs;
    }

    // ---- rescale running output accumulators
#pragma unroll
    for (int t = 0; t < 4; ++t)
#pragma unroll
      for (int r = 0; r < 8; ++r)
        acc[t][r] *= rsc[wave][hi * 8 + r];

    // ---- P fragment (16x32 bf16, A-matrix layout) from LDS
    v16bf ap;
#pragma unroll
    for (int i = 0; i < 8; ++i) {
      ap[i]     = (__bf16)sc[wave][col][hi * 16 + i];
      ap[8 + i] = (__bf16)sc[wave][col][hi * 16 + 8 + i];
    }

    // ---- out += P x V over 4 hd tiles, V from LDS ([d][key])
#pragma unroll
    for (int t = 0; t < 4; ++t) {
      v16us vu = *(const v16us*)&vbuf[buf][(t * 16 + col) * 32 + hi * 16];
      acc[t] = __builtin_amdgcn_wmma_f32_16x16x32_bf16(
          false, ap, false, as_bf(vu), (short)0, acc[t], false, false);
    }
  }

  // ---- epilogue: normalize by l and store f32 [B,S,D]
#pragma unroll
  for (int r = 0; r < 8; ++r) {
    const int M = hi * 8 + r;
    float inv = 1.0f / lrow[wave][M];
    float* orow = out + ((size_t)b * SS + q0 + M) * DDIM + h * HD;
#pragma unroll
    for (int t = 0; t < 4; ++t)
      orow[t * 16 + col] = acc[t][r] * inv;
  }
}

// ---------------------------------------------------------------------------
extern "C" void kernel_launch(void* const* d_in, const int* in_sizes, int n_in,
                              void* d_out, int out_size, void* d_ws,
                              size_t ws_size, hipStream_t stream) {
  (void)in_sizes; (void)n_in; (void)out_size; (void)ws_size;
  const float* x   = (const float*)d_in[0];
  const int*  msk  = (const int*)d_in[1];
  const float* Wq  = (const float*)d_in[2];
  const float* bq  = (const float*)d_in[3];
  const float* Wk  = (const float*)d_in[4];
  const float* bk  = (const float*)d_in[5];
  const float* Wv  = (const float*)d_in[6];
  const float* bv  = (const float*)d_in[7];
  float* out = (float*)d_out;

  // workspace layout (ushorts): Q | K | V^T | x_bf16 | Wq_bf16 | Wk | Wv
  const size_t NQKV = (size_t)BB * HH * SS * HD;   // 4 Mi
  const size_t NX   = (size_t)BSROWS * DDIM;       // 4 Mi
  const size_t NW   = (size_t)DDIM * DDIM;         // 1 Mi
  unsigned short* q   = (unsigned short*)d_ws;
  unsigned short* k   = q + NQKV;
  unsigned short* v   = k + NQKV;
  unsigned short* xb  = v + NQKV;
  unsigned short* wqb = xb + NX;
  unsigned short* wkb = wqb + NW;
  unsigned short* wvb = wkb + NW;

  // ---- prep: bulk f32 -> bf16
  cvt_bf16_kernel<<<(int)(NX / (8 * 256)), 256, 0, stream>>>(x,  xb,  (int)NX);
  cvt_bf16_kernel<<<(int)(NW / (8 * 256)), 256, 0, stream>>>(Wq, wqb, (int)NW);
  cvt_bf16_kernel<<<(int)(NW / (8 * 256)), 256, 0, stream>>>(Wk, wkb, (int)NW);
  cvt_bf16_kernel<<<(int)(NW / (8 * 256)), 256, 0, stream>>>(Wv, wvb, (int)NW);

  // ---- QKV projection
  dim3 g1(BSROWS / 16, DDIM / 512, 3);   // 256 x 2 x 3 blocks, 8 waves each
  qkv_proj_kernel<<<g1, 256, 0, stream>>>(xb, wqb, wkb, wvb,
                                          bq, bk, bv, q, k, v);

  // ---- flash attention
  dim3 g2(SS / 64, HH, BB);              // 32 x 16 x 2 blocks, 4 waves each
  attn_kernel<<<g2, 128, 0, stream>>>(q, k, v, msk, out);
}
